// SparseNeuronAttention_88424786690645
// MI455X (gfx1250) — matrix operand order repaired and checked by reading, not compiled, not run on hardware
//
#include <hip/hip_runtime.h>
#include <hip/hip_bf16.h>

#define BB 4
#define NN 1024
#define DD 1024
#define HH 16
#define HD 64
#define TOPK 64
#define NSTATIC 32

typedef __attribute__((ext_vector_type(16))) _Float16 v16h;
typedef __attribute__((ext_vector_type(8)))  _Float16 v8h;
typedef __attribute__((ext_vector_type(8)))  float    v8f;

union AFrag { v16h v; v8h h[2]; _Float16 e[16]; };
union CFrag { v8f  v; float f[8]; };

// ---------------- WMMA helpers (gfx1250 wave32, 16x16x32 f16 -> f32) ----------------

__device__ __forceinline__ v8f wmma_f16(v16h a, v16h b, v8f c) {
  // 8 args: (neg_a, A, neg_b, B, c_mod, C, reuse_a, reuse_b)
  return __builtin_amdgcn_wmma_f32_16x16x32_f16(false, a, false, b, (short)0, c,
                                                false, false);
}

// A-matrix 16x32 f16 fragment (row-major source):
// lane holds row (lane&15); element j<8 -> K = c0+j, j>=8 -> K = 16+c0+(j-8),
// c0 = (lane<16)?0:8.  Two aligned 16B loads.
__device__ __forceinline__ v16h load_a_frag(const _Float16* __restrict__ rowptr,
                                            int kc, int lane) {
  AFrag a;
  int c0 = (lane < 16) ? 0 : 8;
  a.h[0] = *(const v8h*)(rowptr + kc + c0);
  a.h[1] = *(const v8h*)(rowptr + kc + 16 + c0);
  return a.v;
}

// B-matrix 32x16 f16 fragment where B[k][n] = M[n][k] (M row-major):
// lane holds column (lane&15); element j<8 -> K = c0+j, j>=8 -> K = 16+c0+(j-8), c0 = (lane<16)?0:8.
__device__ __forceinline__ v16h load_b_frag(const _Float16* __restrict__ colrowptr,
                                            int kc, int lane) {
  AFrag b;
  int c0 = (lane < 16) ? 0 : 8;
  b.h[0] = *(const v8h*)(colrowptr + kc + c0);
  b.h[1] = *(const v8h*)(colrowptr + kc + 16 + c0);
  return b.v;
}

__device__ __forceinline__ unsigned f2o(float f) {   // order-preserving f32->u32
  unsigned u = __float_as_uint(f);
  return (u & 0x80000000u) ? ~u : (u | 0x80000000u);
}

// ---------------- elementwise f32 -> f16 ----------------

__global__ void cvt_f32_f16(const float* __restrict__ s, _Float16* __restrict__ d, int n) {
  int i = blockIdx.x * blockDim.x + threadIdx.x;
  if (i < n) d[i] = (_Float16)s[i];
}

// ---------------- WMMA GEMM:  C[m][n] = sum_k A[m][k] * W[n][k] + bias[n] ----------
// 64x32 tile per wave: 4x2 C-fragments (64 VGPRs), 8 WMMA per 32-wide k-chunk.
// ~21 FLOP/B from L2, ~140 VGPRs total -> no spills, room to double-buffer.
// mode 0: Q -> f16 head layout, scaled 1/sqrt(HD);  mode 1: K -> f16 head layout
// mode 2: V -> f16 transposed head layout [bh][hd][tok];  mode 3: f32 row-major out

__global__ void __launch_bounds__(128)
gemm64x32(const _Float16* __restrict__ A, const _Float16* __restrict__ W,
          const float* __restrict__ bias, void* __restrict__ out,
          int Kd, int mode) {
  int lane = threadIdx.x & 31;
  int wave = threadIdx.x >> 5;
  int tile = blockIdx.x * 4 + wave;        // 2048 wave-tiles: 64 (M) x 32 (N)
  int m0 = (tile >> 5) << 6;               // 0..4032
  int n0 = (tile & 31) << 5;               // 0..992

  const _Float16* arow[4];
  const _Float16* brow[2];
#pragma unroll
  for (int mi = 0; mi < 4; ++mi)
    arow[mi] = A + (size_t)(m0 + (mi << 4) + (lane & 15)) * DD;
#pragma unroll
  for (int ni = 0; ni < 2; ++ni)
    brow[ni] = W + (size_t)(n0 + (ni << 4) + (lane & 15)) * DD;

  CFrag acc[4][2];
#pragma unroll
  for (int mi = 0; mi < 4; ++mi)
#pragma unroll
    for (int ni = 0; ni < 2; ++ni)
#pragma unroll
      for (int j = 0; j < 8; ++j) acc[mi][ni].f[j] = 0.0f;

#pragma unroll 1
  for (int kc = 0; kc < Kd; kc += 32) {
    // prefetch next cachelines of the streaming rows (global_prefetch_b8)
    __builtin_prefetch(arow[0] + kc + 128, 0, 1);
    __builtin_prefetch(brow[0] + kc + 128, 0, 1);
    v16h a[4], b[2];
#pragma unroll
    for (int mi = 0; mi < 4; ++mi) a[mi] = load_a_frag(arow[mi], kc, lane);
#pragma unroll
    for (int ni = 0; ni < 2; ++ni) b[ni] = load_b_frag(brow[ni], kc, lane);
#pragma unroll
    for (int mi = 0; mi < 4; ++mi)
#pragma unroll
      for (int ni = 0; ni < 2; ++ni)
        acc[mi][ni].v = wmma_f16(a[mi], b[ni], acc[mi][ni].v);
  }

#pragma unroll
  for (int ni = 0; ni < 2; ++ni) {
    int n = n0 + (ni << 4) + (lane & 15);
    float bval = bias[n];
    int h = n >> 6, hd = n & 63;
#pragma unroll
    for (int mi = 0; mi < 4; ++mi) {
#pragma unroll
      for (int j = 0; j < 8; ++j) {
        int m    = m0 + (mi << 4) + j + ((lane < 16) ? 0 : 8);
        int bidx = m >> 10;
        int tok  = m & (NN - 1);
        float val = acc[mi][ni].f[j] + bval;
        if (mode == 0) {
          ((_Float16*)out)[((size_t)(bidx * HH + h) * NN + tok) * HD + hd] =
              (_Float16)(val * 0.125f);                   // 1/sqrt(64)
        } else if (mode == 1) {
          ((_Float16*)out)[((size_t)(bidx * HH + h) * NN + tok) * HD + hd] =
              (_Float16)val;
        } else if (mode == 2) {
          ((_Float16*)out)[((size_t)(bidx * HH + h) * HD + hd) * NN + tok] =
              (_Float16)val;
        } else {
          ((float*)out)[(size_t)m * DD + n] = val;
        }
      }
    }
  }
}

// ---------------- overwrite first NUM_STATIC key rows with static keys ----------------

__global__ void static_kv(const float* __restrict__ sk, _Float16* __restrict__ Kh) {
  int i = blockIdx.x * blockDim.x + threadIdx.x;      // over B*NSTATIC*DD
  if (i >= BB * NSTATIC * DD) return;
  int n = i & (DD - 1);
  int s = (i >> 10) & (NSTATIC - 1);
  int b = i >> 15;
  int h = n >> 6, hd = n & 63;
  Kh[((size_t)(b * HH + h) * NN + s) * HD + hd] = (_Float16)sk[s * DD + n];
}

// ---------------- fused scores + top-k softmax + attn@V ----------------
// grid = B*H*(N/64) = 1024 blocks, 128 threads (4 wave32, 16 queries each).
// Dynamic LDS: 64x1024 f32 score tile (256KB) + 64 f32 recip sums — uses the
// CDNA5 320KB-per-WGP LDS (not possible on CDNA4's 64KB).

__global__ void attn64(const _Float16* __restrict__ Qh, const _Float16* __restrict__ Kh,
                       const _Float16* __restrict__ Vt, _Float16* __restrict__ AO) {
  extern __shared__ char smemraw[];
  float* S        = (float*)smemraw;                          // [64][1024]
  float* rowRecip = (float*)(smemraw + 64 * 1024 * 4);        // [64]

  int lane = threadIdx.x & 31;
  int wave = threadIdx.x >> 5;
  int bh = blockIdx.x >> 4;       // b*H + h
  int qt = blockIdx.x & 15;       // query tile within head
  int b = bh >> 4, h = bh & 15;

  const _Float16* Qbh = Qh + (size_t)bh * NN * HD;
  const _Float16* Kbh = Kh + (size_t)bh * NN * HD;
  const _Float16* Vtb = Vt + (size_t)bh * HD * NN;

  int rbase = wave << 4;                 // this wave's rows in the 64-row tile
  int q0    = (qt << 6) + rbase;         // global token index of row 0

  // ---- phase 1: scores S[16 x 1024] = Qtile (pre-scaled) @ K^T ----
  const _Float16* qrow = Qbh + (size_t)(q0 + (lane & 15)) * HD;
  v16h aq0 = load_a_frag(qrow, 0, lane);
  v16h aq1 = load_a_frag(qrow, 32, lane);

  for (int kb = 0; kb < NN; kb += 16) {
    const _Float16* krow = Kbh + (size_t)(kb + (lane & 15)) * HD;
    CFrag c;
#pragma unroll
    for (int j = 0; j < 8; ++j) c.f[j] = 0.0f;
    c.v = wmma_f16(aq0, load_b_frag(krow, 0, lane), c.v);
    c.v = wmma_f16(aq1, load_b_frag(krow, 32, lane), c.v);
#pragma unroll
    for (int j = 0; j < 8; ++j) {
      int row = rbase + j + ((lane < 16) ? 0 : 8);
      S[row * 1024 + kb + (lane & 15)] = c.f[j];
    }
  }
  __syncthreads();

  // ---- phase 2: per-row radix-select top-64 threshold, exp weights in place ----
  for (int rr = 0; rr < 16; ++rr) {
    int row = rbase + rr;
    float fv[32]; unsigned uu[32];
    float m = -1e30f;
#pragma unroll
    for (int i = 0; i < 32; ++i) {
      fv[i] = S[row * 1024 + i * 32 + lane];
      uu[i] = f2o(fv[i]);
      m = fmaxf(m, fv[i]);
    }
#pragma unroll
    for (int off = 16; off > 0; off >>= 1) m = fmaxf(m, __shfl_xor(m, off, 32));

    unsigned t = 0u;
    for (int bit = 31; bit >= 0; --bit) {         // radix select K-th largest bucket
      unsigned cand = t | (1u << bit);
      int cnt = 0;
#pragma unroll
      for (int i = 0; i < 32; ++i) cnt += (uu[i] >= cand) ? 1 : 0;
#pragma unroll
      for (int off = 16; off > 0; off >>= 1) cnt += __shfl_xor(cnt, off, 32);
      if (cnt >= TOPK) t = cand;
    }

    float lsum = 0.0f;
#pragma unroll
    for (int i = 0; i < 32; ++i) {
      float w = (uu[i] >= t) ? __expf(fv[i] - m) : 0.0f;   // masked softmax numerator
      S[row * 1024 + i * 32 + lane] = w;
      lsum += w;
    }
#pragma unroll
    for (int off = 16; off > 0; off >>= 1) lsum += __shfl_xor(lsum, off, 32);
    if (lane == 0) rowRecip[row] = 1.0f / lsum;
  }
  __syncthreads();

  // ---- phase 3: O[16 x 64] = attn[16 x 1024] @ V   (B from transposed Vt) ----
  CFrag o[4];
#pragma unroll
  for (int tq = 0; tq < 4; ++tq)
#pragma unroll
    for (int j = 0; j < 8; ++j) o[tq].f[j] = 0.0f;

  for (int kc = 0; kc < NN; kc += 32) {
    AFrag a;
    int c0 = (lane < 16) ? 0 : 8;
    const float* srow = S + (rbase + (lane & 15)) * 1024 + kc;
#pragma unroll
    for (int j = 0; j < 8; ++j) a.e[j]     = (_Float16)srow[c0 + j];
#pragma unroll
    for (int j = 0; j < 8; ++j) a.e[8 + j] = (_Float16)srow[16 + c0 + j];
#pragma unroll
    for (int tq = 0; tq < 4; ++tq) {
      const _Float16* vrow = Vtb + (size_t)((tq << 4) + (lane & 15)) * NN;
      o[tq].v = wmma_f16(a.v, load_b_frag(vrow, kc, lane), o[tq].v);
    }
  }

  // ---- phase 4: normalize by softmax sum, store f16 [B,N,D] head-interleaved ----
#pragma unroll
  for (int tq = 0; tq < 4; ++tq) {
#pragma unroll
    for (int j = 0; j < 8; ++j) {
      int row = rbase + j + ((lane < 16) ? 0 : 8);
      int tok = (qt << 6) + row;
      float val = o[tq].f[j] * rowRecip[row];
      AO[(size_t)(b * NN + tok) * DD + (h << 6) + (tq << 4) + (lane & 15)] =
          (_Float16)val;
    }
  }
}

// ---------------- launch ----------------

extern "C" void kernel_launch(void* const* d_in, const int* in_sizes, int n_in,
                              void* d_out, int out_size, void* d_ws, size_t ws_size,
                              hipStream_t stream) {
  (void)in_sizes; (void)n_in; (void)out_size; (void)ws_size;
  const float* x  = (const float*)d_in[0];
  const float* Wq = (const float*)d_in[1];
  const float* bq = (const float*)d_in[2];
  const float* Wk = (const float*)d_in[3];
  const float* bk = (const float*)d_in[4];
  const float* Wv = (const float*)d_in[5];
  const float* bv = (const float*)d_in[6];
  const float* Wo = (const float*)d_in[7];
  const float* bo = (const float*)d_in[8];
  const float* sk = (const float*)d_in[9];

  char* ws = (char*)d_ws;                       // 48MB layout (MB-aligned)
  _Float16* xh  = (_Float16*)(ws + ((size_t)0  << 20));
  _Float16* Wqh = (_Float16*)(ws + ((size_t)8  << 20));
  _Float16* Wkh = (_Float16*)(ws + ((size_t)10 << 20));
  _Float16* Wvh = (_Float16*)(ws + ((size_t)12 << 20));
  _Float16* Woh = (_Float16*)(ws + ((size_t)14 << 20));
  _Float16* Qh  = (_Float16*)(ws + ((size_t)16 << 20));
  _Float16* Kh  = (_Float16*)(ws + ((size_t)24 << 20));
  _Float16* Vt  = (_Float16*)(ws + ((size_t)32 << 20));
  _Float16* AO  = (_Float16*)(ws + ((size_t)40 << 20));

  const int nX = BB * NN * DD;     // 4M
  const int nW = DD * DD;          // 1M
  cvt_f32_f16<<<nX / 256, 256, 0, stream>>>(x,  xh,  nX);
  cvt_f32_f16<<<nW / 256, 256, 0, stream>>>(Wq, Wqh, nW);
  cvt_f32_f16<<<nW / 256, 256, 0, stream>>>(Wk, Wkh, nW);
  cvt_f32_f16<<<nW / 256, 256, 0, stream>>>(Wv, Wvh, nW);
  cvt_f32_f16<<<nW / 256, 256, 0, stream>>>(Wo, Woh, nW);

  // QKV projections: 2048 wave-tiles of 64x32 -> 512 blocks of 4 waves
  gemm64x32<<<512, 128, 0, stream>>>(xh, Wqh, bq, Qh, DD, 0);
  gemm64x32<<<512, 128, 0, stream>>>(xh, Wkh, bk, Kh, DD, 1);
  gemm64x32<<<512, 128, 0, stream>>>(xh, Wvh, bv, Vt, DD, 2);

  static_kv<<<(BB * NSTATIC * DD) / 256, 256, 0, stream>>>(sk, Kh);

  size_t smem = (size_t)64 * 1024 * 4 + 256;   // 256KB score tile + recips
  attn64<<<BB * HH * (NN / 64), 128, smem, stream>>>(Qh, Kh, Vt, AO);

  // output projection -> f32 d_out
  gemm64x32<<<512, 128, 0, stream>>>(AO, Woh, bo, (float*)d_out, DD, 3);
}
